// MemAttention_15736760172669
// MI455X (gfx1250) — compile-verified, matrix operand order checked by reading, not run on hardware
//
#include <hip/hip_runtime.h>

typedef __attribute__((ext_vector_type(16))) _Float16 v16h;
typedef __attribute__((ext_vector_type(8)))  _Float16 v8h;
typedef __attribute__((ext_vector_type(8)))  float    v8f;

constexpr int Bc  = 2;
constexpr int Hc  = 8;
constexpr int Lc  = 2048;
constexpr int Ec  = 64;
constexpr int KVc = 2560;   // L + MEM
constexpr int WIN = 512;    // mem_len (causal window offset)

constexpr int KP = 72;      // kbuf row pitch (halves), 144B = 16B-aligned
constexpr int VP = 40;      // vbuf row pitch (halves), 80B  = 16B-aligned
constexpr int PP = 40;      // pbuf row pitch (halves)

constexpr size_t PE16_BYTES = (size_t)Hc * KVc * Ec * 2;   // f16 pe cache in d_ws

__device__ __forceinline__ v8f wmma16(v16h a, v16h b, v8f c) {
    // D = A(16x32 f16) * B(32x16 f16) + C(16x16 f32)
    return __builtin_amdgcn_wmma_f32_16x16x32_f16(
        false, a, false, b, (short)0, c, false, false);
}

// f16 A/B fragment from a row of 64 fp32 values in global memory.
// Per-lane pattern (ISA 16-bit 16x32 layout): halves 0..7 = K = ko+0..7,
// halves 8..15 = K = 16+ko+0..7, where ko = (lane>=16 ? 8 : 0).
__device__ __forceinline__ v16h ld_frag_f32(const float* __restrict__ rowp, int lane) {
    const int ko = ((lane >> 4) & 1) * 8;
    const float* p = rowp + ko;
    v16h r;
#pragma unroll
    for (int i = 0; i < 8; ++i) r[i] = (_Float16)p[i];
#pragma unroll
    for (int i = 0; i < 8; ++i) r[8 + i] = (_Float16)p[16 + i];
    return r;
}

// Same fragment pattern from an f16 row (LDS or global): two 16B loads, no cvt.
__device__ __forceinline__ v16h ld_frag_h16(const _Float16* rowp, int lane) {
    const int ko = ((lane >> 4) & 1) * 8;
    v8h lo = *(const v8h*)(rowp + ko);
    v8h hi = *(const v8h*)(rowp + 16 + ko);
    return __builtin_shufflevector(lo, hi, 0, 1, 2, 3, 4, 5, 6, 7,
                                   8, 9, 10, 11, 12, 13, 14, 15);
}

// U tile: U[mm, n] = q[m0+mm] . pe[tbase + n]   (16x16 f32 via 2 WMMAs)
template <bool PE16>
__device__ __forceinline__ v8f compute_u(const float* __restrict__ peh,
                                         const _Float16* __restrict__ peh16,
                                         int tbase, int lane, v16h qa0, v16h qa1) {
    int tr = tbase + (lane & 15);
    tr = tr < 0 ? 0 : (tr >= KVc ? KVc - 1 : tr);   // clamped rows feed masked slots only
    v16h b0, b1;
    if constexpr (PE16) {
        const _Float16* rowp = peh16 + (size_t)tr * Ec;
        b0 = ld_frag_h16(rowp, lane);
        b1 = ld_frag_h16(rowp + 32, lane);
    } else {
        const float* rowp = peh + (size_t)tr * Ec;
        b0 = ld_frag_f32(rowp, lane);
        b1 = ld_frag_f32(rowp + 32, lane);
    }
    v8f u = {};
    u = wmma16(qa0, b0, u);
    u = wmma16(qa1, b1, u);
    return u;
}

// One-time pe fp32 -> f16 cache (2.6 MB), amortized across all (b, j-step) reuse.
__global__ __launch_bounds__(256) void pe_cvt_kernel(const float* __restrict__ pe,
                                                     _Float16* __restrict__ pe16) {
    const size_t base = ((size_t)blockIdx.x * 256 + threadIdx.x) * 8;
    if (base >= (size_t)Hc * KVc * Ec) return;
    v8h hv;
#pragma unroll
    for (int i = 0; i < 8; ++i) hv[i] = (_Float16)pe[base + i];
    *(v8h*)(pe16 + base) = hv;
}

template <bool PE16>
__global__ __launch_bounds__(256) void
MemAttention_xl_kernel(const float* __restrict__ q, const float* __restrict__ k,
                       const float* __restrict__ v, const float* __restrict__ pe,
                       const _Float16* __restrict__ pe16g, float* __restrict__ out) {
    __shared__ __align__(16) _Float16 kbuf[2][32 * KP];    // K chunk, row-major f16 (dbl-buf)
    __shared__ __align__(16) _Float16 vbuf[2][64 * VP];    // V chunk, TRANSPOSED [e][kk]
    __shared__ __align__(16) _Float16 pbuf[8][16 * PP];    // per-wave prob transpose

    const int tid  = threadIdx.x;
    const int lane = tid & 31;
    const int w    = tid >> 5;

    const int tile = blockIdx.x & 15;    // 16 row-tiles of 128
    const int bh   = blockIdx.x >> 4;    // 0..15
    const int h    = bh & 7;
    const int b    = bh >> 3;

    const int row0 = tile * 128;
    const int m0   = row0 + w * 16;
    const int jend = row0 + 640;         // (127+512+1) rounded: always <= KV, mult of 32

    const float*    kb   = k  + (size_t)bh * KVc * Ec;
    const float*    vb   = v  + (size_t)bh * KVc * Ec;
    const float*    qb   = q  + (size_t)bh * (size_t)Lc * Ec;
    const float*    peh  = pe + (size_t)h * KVc * Ec;
    const _Float16* peh16 = PE16 ? (pe16g + (size_t)h * KVc * Ec) : nullptr;

    // Q A-fragments, held in registers for the whole sweep (also reused for pos GEMM)
    const int qrow = m0 + (lane & 15);
    const v16h qa0 = ld_frag_f32(qb + (size_t)qrow * Ec, lane);
    const v16h qa1 = ld_frag_f32(qb + (size_t)qrow * Ec + 32, lane);

    v8f acc0 = {}, acc1 = {}, acc2 = {}, acc3 = {};   // out accumulators, E tiles 0..3
    float rs[8];
#pragma unroll
    for (int r = 0; r < 8; ++r) rs[r] = 0.0f;

    // rolling U tile: T0(j0+32) == T2(j0); seed with tile at T0(j0=0)
    v8f u2 = compute_u<PE16>(peh, peh16, -m0 + (Lc - 1) - 16, lane, qa0, qa1);

    _Float16* pb = pbuf[w];
    const int n  = lane & 15;            // C-layout column
    const int g8 = (lane >> 1) & 8;      // 8 * (lane>>4)

    for (int j0 = 0; j0 < jend; j0 += 32) {
        const int sel = (j0 >> 5) & 1;
        {   // cooperative stage of K,V (32 rows x 64 feats, fp32 -> f16)
            const int lr = tid >> 3;          // 0..31  local kv row
            const int fb = (tid & 7) * 8;     // 0..56  feature base
            const float* kr = kb + (size_t)(j0 + lr) * Ec + fb;
            const float* vr = vb + (size_t)(j0 + lr) * Ec + fb;
            v8h kh;
#pragma unroll
            for (int i = 0; i < 8; ++i) kh[i] = (_Float16)kr[i];
            *(v8h*)&kbuf[sel][lr * KP + fb] = kh;
#pragma unroll
            for (int i = 0; i < 8; ++i)        // transpose-on-store for V
                vbuf[sel][(fb + i) * VP + lr] = (_Float16)vr[i];
            if (j0 + 64 <= jend) {
                __builtin_prefetch(kr + 32 * Ec, 0, 1);
                __builtin_prefetch(vr + 32 * Ec, 0, 1);
            }
        }
        __syncthreads();   // single barrier/step: dbl-buffer protects next-step writes

        // ---- QK^T: two 16x16 score tiles (4 WMMAs) ----
        const _Float16* kbl = kbuf[sel];
        v8f s0 = {}, s1 = {};
        s0 = wmma16(qa0, ld_frag_h16(&kbl[n * KP], lane), s0);
        s0 = wmma16(qa1, ld_frag_h16(&kbl[n * KP + 32], lane), s0);
        s1 = wmma16(qa0, ld_frag_h16(&kbl[(16 + n) * KP], lane), s1);
        s1 = wmma16(qa1, ld_frag_h16(&kbl[(16 + n) * KP + 32], lane), s1);

        // ---- positional GEMM, t-aligned U tiles (rolling; 4 WMMAs/step) ----
        const int T0 = j0 - m0 + (Lc - 1) - 16;
        v8f u0 = u2;
        v8f u1 = compute_u<PE16>(peh, peh16, T0 + 16, lane, qa0, qa1);
        u2      = compute_u<PE16>(peh, peh16, T0 + 32, lane, qa0, qa1);

        // ---- rel-shift select + mask + exp; stash probs (f16) for transpose ----
#pragma unroll
        for (int r = 0; r < 8; ++r) {
            const int mm  = r + g8;                       // C-layout row in tile
            const int src = (lane & 16) | ((n - mm) & 15);
            const float a0 = __shfl(u0[r], src, 32);
            const float a1 = __shfl(u1[r], src, 32);
            const float a2 = __shfl(u2[r], src, 32);
            const bool low = n < mm;                      // which diagonal half
            const float pos0 = low ? a0 : a1;             // t = T0+16 + (n-mm)
            const float pos1 = low ? a1 : a2;             // t = T0+32 + (n-mm)
            const int m = m0 + mm;
            const float sc0 = (s0[r] + pos0) * 0.125f;
            const float sc1 = (s1[r] + pos1) * 0.125f;
            const float p0 = (j0 + n)      <= m + WIN ? __expf(sc0) : 0.0f;
            const float p1 = (j0 + 16 + n) <= m + WIN ? __expf(sc1) : 0.0f;
            rs[r] += p0 + p1;
            pb[mm * PP + n]      = (_Float16)p0;
            pb[mm * PP + 16 + n] = (_Float16)p1;
        }

        // ---- P (16x32) as A-fragment via per-wave LDS transpose; PV (4 WMMAs) ----
        const v16h ap = ld_frag_h16(&pb[(lane & 15) * PP], lane);
        const _Float16* vbl = vbuf[sel];
        acc0 = wmma16(ap, ld_frag_h16(&vbl[(0  + n) * VP], lane), acc0);
        acc1 = wmma16(ap, ld_frag_h16(&vbl[(16 + n) * VP], lane), acc1);
        acc2 = wmma16(ap, ld_frag_h16(&vbl[(32 + n) * VP], lane), acc2);
        acc3 = wmma16(ap, ld_frag_h16(&vbl[(48 + n) * VP], lane), acc3);
    }

    // ---- row-sum reduction (16-lane groups) and normalized store ----
    float inv[8];
#pragma unroll
    for (int r = 0; r < 8; ++r) {
        float s = rs[r];
        s += __shfl_xor(s, 1, 32);
        s += __shfl_xor(s, 2, 32);
        s += __shfl_xor(s, 4, 32);
        s += __shfl_xor(s, 8, 32);
        inv[r] = 1.0f / s;
    }
#pragma unroll
    for (int r = 0; r < 8; ++r) {
        const int m = m0 + r + g8;
        const size_t base = ((size_t)b * Lc + m) * (size_t)(Hc * Ec) + h * Ec + n;
        out[base +  0] = acc0[r] * inv[r];
        out[base + 16] = acc1[r] * inv[r];
        out[base + 32] = acc2[r] * inv[r];
        out[base + 48] = acc3[r] * inv[r];
    }
}

extern "C" void kernel_launch(void* const* d_in, const int* in_sizes, int n_in,
                              void* d_out, int out_size, void* d_ws, size_t ws_size,
                              hipStream_t stream) {
    (void)in_sizes; (void)n_in; (void)out_size;
    const float* q  = (const float*)d_in[0];
    const float* k  = (const float*)d_in[1];
    const float* v  = (const float*)d_in[2];
    // d_in[3] = attn_mask (scalar 0), d_in[5..7] = kv_len/mem_len/lmem_len (static here)
    const float* pe = (const float*)d_in[4];
    float* out = (float*)d_out;

    dim3 grid(Bc * Hc * (Lc / 128));   // 256 blocks
    dim3 block(256);                   // 8 waves (wave32)

    if (d_ws != nullptr && ws_size >= PE16_BYTES) {
        _Float16* pe16 = (_Float16*)d_ws;
        const size_t n8 = (size_t)Hc * KVc * Ec / 8;           // elements / 8 per thread
        pe_cvt_kernel<<<dim3((unsigned)((n8 + 255) / 256)), dim3(256), 0, stream>>>(pe, pe16);
        MemAttention_xl_kernel<true><<<grid, block, 0, stream>>>(q, k, v, pe, pe16, out);
    } else {
        MemAttention_xl_kernel<false><<<grid, block, 0, stream>>>(q, k, v, pe, nullptr, out);
    }
}
